// PerformerSimple_22290880266616
// MI455X (gfx1250) — matrix operand order repaired and checked by reading, not compiled, not run on hardware
//
#include <hip/hip_runtime.h>
#include <math.h>

// ---------------------------------------------------------------------------
// Performer linear attention for MI455X (gfx1250, wave32, WMMA).
// All GEMMs use v_wmma_f32_16x16x32_bf16 (fp32 accumulate).
// b=8, c=EMB=512, t=4096, m=256.
// Roofline: ~34 GFLOP GEMM, ~256MB HBM floor (~11us @ 23.3TB/s) -> bf16 WMMA
// keeps compute (~7us) at parity with memory; fp32 VALU would be ~100x slower.
// ---------------------------------------------------------------------------

typedef __attribute__((ext_vector_type(16))) __bf16 v16bf;
typedef __attribute__((ext_vector_type(8)))  float  v8f;

#define B_     8
#define C_     512
#define T_     4096
#define M_     256
#define EPS_   1e-8f
#define LNEPS_ 1e-5f

// Convert two 8-float chunks (A-matrix K pattern) into one v16bf A operand.
// ISA 16-bit A 16x32 layout: lanes 0-15 hold K {0..7,16..23}, lanes 16-31
// hold K {8..15,24..31} for their M row.
__device__ __forceinline__ v16bf cvt_a16(const float* pa, const float* pb) {
    float4 x0 = *(const float4*)pa, x1 = *(const float4*)(pa + 4);
    float4 y0 = *(const float4*)pb, y1 = *(const float4*)(pb + 4);
    v16bf a;
    a[0]=(__bf16)x0.x; a[1]=(__bf16)x0.y; a[2]=(__bf16)x0.z; a[3]=(__bf16)x0.w;
    a[4]=(__bf16)x1.x; a[5]=(__bf16)x1.y; a[6]=(__bf16)x1.z; a[7]=(__bf16)x1.w;
    a[8]=(__bf16)y0.x; a[9]=(__bf16)y0.y; a[10]=(__bf16)y0.z; a[11]=(__bf16)y0.w;
    a[12]=(__bf16)y1.x; a[13]=(__bf16)y1.y; a[14]=(__bf16)y1.z; a[15]=(__bf16)y1.w;
    return a;
}

__device__ __forceinline__ v16bf bf16_a16(const __bf16* pa, const __bf16* pb) {
    v16bf a;
#pragma unroll
    for (int j = 0; j < 8; ++j) { a[j] = pa[j]; a[8 + j] = pb[j]; }
    return a;
}

// -------------------- 0) xd[b][t] = 0.5 * sum_c x[c][t]^2 -------------------
__global__ void xd_kernel(const float* __restrict__ X, float* __restrict__ xd) {
    int idx = blockIdx.x * 256 + threadIdx.x;          // b*T_
    int b = idx >> 12, t = idx & (T_ - 1);
    const float* p = X + (size_t)b * C_ * T_ + t;
    float s = 0.f;
#pragma unroll 8
    for (int c = 0; c < C_; ++c) { float v = p[(size_t)c * T_]; s += v * v; }
    xd[idx] = 0.5f * s;
}

// ---- 1) features: P^T[m][t] = W[m,c] x X[c,t]; p = exp(wtx - xd)/sqrt(m) ---
// MODE 0: store as qpT [b][m][t];  MODE 1: store as kp [b][t][m]
template <int MODE>
__global__ void feat_kernel(const float* __restrict__ X,   // [b][C_][T_]
                            const float* __restrict__ W,   // [M_][C_]
                            const float* __restrict__ xd,  // [b][T_]
                            __bf16* __restrict__ out) {
    __shared__ __align__(32) __bf16 sB[32 * 64];
    const int lane = threadIdx.x & 31;
    const int wave = threadIdx.x >> 5;
    const int lo = lane & 15, hi = lane >> 4;
    const int b = blockIdx.z;
    const int tbase = blockIdx.x * 64;
    const int m0 = (blockIdx.y * 8 + wave) * 16;

    const float* Xb = X + (size_t)b * C_ * T_;
    const float* wr = W + (size_t)(m0 + lo) * C_;

    v8f acc[4] = {};
    for (int k0 = 0; k0 < C_; k0 += 32) {
        __syncthreads();
        {   // cooperative LDS stage: B chunk rows k0..k0+31 (c), cols tbase..+63
            int r = threadIdx.x >> 3;             // 0..31
            int cg = (threadIdx.x & 7) * 8;       // 0,8,..56
            const float* src = Xb + (size_t)(k0 + r) * T_ + tbase + cg;
            float4 u = *(const float4*)src, w2 = *(const float4*)(src + 4);
            __bf16* d = &sB[r * 64 + cg];
            d[0]=(__bf16)u.x; d[1]=(__bf16)u.y; d[2]=(__bf16)u.z; d[3]=(__bf16)u.w;
            d[4]=(__bf16)w2.x; d[5]=(__bf16)w2.y; d[6]=(__bf16)w2.z; d[7]=(__bf16)w2.w;
        }
        __syncthreads();
        v16bf a = cvt_a16(wr + k0 + hi * 8, wr + k0 + 16 + hi * 8);
#pragma unroll
        for (int j = 0; j < 4; ++j) {
            v16bf bm;
#pragma unroll
            for (int s = 0; s < 16; ++s) bm[s] = sB[(hi * 8 + (s & 7) + (s >> 3) * 16) * 64 + j * 16 + lo];
            acc[j] = __builtin_amdgcn_wmma_f32_16x16x32_bf16(
                false, a, false, bm, (short)0, acc[j], false, false);
        }
    }
    const float inv_sqrt_m = 0.0625f;  // 1/sqrt(256)
#pragma unroll
    for (int j = 0; j < 4; ++j) {
        int tCol = tbase + j * 16 + lo;
        float xdv = xd[b * T_ + tCol];
#pragma unroll
        for (int i = 0; i < 8; ++i) {
            int mRow = m0 + i + (hi ? 8 : 0);
            float val = __expf(acc[j][i] - xdv) * inv_sqrt_m;
            if (MODE == 0)
                out[((size_t)b * M_ + mRow) * T_ + tCol] = (__bf16)val;
            else
                out[((size_t)b * T_ + tCol) * M_ + mRow] = (__bf16)val;
        }
    }
}

// ---- 2a) partial ksum over t-chunks: part[b][ch][m] = sum_{t in chunk} kp --
__global__ void ksum_part_kernel(const __bf16* __restrict__ KP,
                                 float* __restrict__ part) {
    int b = blockIdx.x, ch = blockIdx.y, m = threadIdx.x;   // 8 x 16 blocks
    const __bf16* p = KP + (size_t)b * T_ * M_ + (size_t)ch * 256 * M_ + m;
    float s = 0.f;
#pragma unroll 8
    for (int t = 0; t < 256; ++t) s += (float)p[(size_t)t * M_];
    part[((size_t)b * 16 + ch) * M_ + m] = s;
}

// ---- 2b) combine partials: ksum[b][m] = sum_ch part[b][ch][m] --------------
__global__ void ksum_final_kernel(const float* __restrict__ part,
                                  float* __restrict__ ksum) {
    int b = blockIdx.x, m = threadIdx.x;
    const float* p = part + (size_t)b * 16 * M_ + m;
    float s = 0.f;
#pragma unroll
    for (int ch = 0; ch < 16; ++ch) s += p[(size_t)ch * M_];
    ksum[b * M_ + m] = s;
}

// -------------------- 3) D[b][t] = sum_m qpT[m][t] * ksum[m] ----------------
__global__ void d_kernel(const __bf16* __restrict__ QPT,
                         const float* __restrict__ ksum, float* __restrict__ Dv) {
    int idx = blockIdx.x * 256 + threadIdx.x;          // b*T_
    int b = idx >> 12, t = idx & (T_ - 1);
    const __bf16* p = QPT + (size_t)b * M_ * T_ + t;
    const float* ks = ksum + b * M_;
    float s = 0.f;
#pragma unroll 8
    for (int m = 0; m < M_; ++m) s += (float)p[(size_t)m * T_] * ks[m];
    Dv[idx] = s;
}

// ----------- 4) kptv[c][m] = sum_t V[c][t] * kp[t][m]  (K = 4096) -----------
// Per wave: 32(c) x 64(m) tile -> 8 accumulators; B loads shared by 2 A tiles.
__global__ void kptv_kernel(const float* __restrict__ V,    // [b][C_][T_]
                            const __bf16* __restrict__ KP,  // [b][T_][M_]
                            __bf16* __restrict__ KPTV) {    // [b][C_][M_]
    const int lane = threadIdx.x & 31;
    const int wave = threadIdx.x >> 5;
    const int lo = lane & 15, hi = lane >> 4;
    const int b = blockIdx.z;
    int job = blockIdx.x * 8 + wave;    // 0..63
    int c0 = (job >> 2) * 32;           // 16 c-blocks of 32
    int mb = (job & 3) * 64;            // 4 m-chunks of 64
    const float* vr0 = V + ((size_t)b * C_ + c0 + lo) * T_;
    const float* vr1 = vr0 + (size_t)16 * T_;
    const __bf16* KPb = KP + (size_t)b * T_ * M_;

    v8f acc0[4] = {}, acc1[4] = {};
    for (int k0 = 0; k0 < T_; k0 += 32) {
        v16bf a0 = cvt_a16(vr0 + k0 + hi * 8, vr0 + k0 + 16 + hi * 8);
        v16bf a1 = cvt_a16(vr1 + k0 + hi * 8, vr1 + k0 + 16 + hi * 8);
#pragma unroll
        for (int j = 0; j < 4; ++j) {
            v16bf bm;
#pragma unroll
            for (int s = 0; s < 16; ++s) bm[s] = KPb[(size_t)(k0 + hi * 8 + (s & 7) + (s >> 3) * 16) * M_ + mb + j * 16 + lo];
            acc0[j] = __builtin_amdgcn_wmma_f32_16x16x32_bf16(
                false, a0, false, bm, (short)0, acc0[j], false, false);
            acc1[j] = __builtin_amdgcn_wmma_f32_16x16x32_bf16(
                false, a1, false, bm, (short)0, acc1[j], false, false);
        }
    }
#pragma unroll
    for (int j = 0; j < 4; ++j)
#pragma unroll
        for (int i = 0; i < 8; ++i) {
            int cRow = c0 + i + (hi ? 8 : 0);
            int mCol = mb + j * 16 + lo;
            KPTV[((size_t)b * C_ + cRow) * M_ + mCol] = (__bf16)acc0[j][i];
            KPTV[((size_t)b * C_ + cRow + 16) * M_ + mCol] = (__bf16)acc1[j][i];
        }
}

// ---- 5) y^T[c][t] = (kptv[c][:] . qpT[:][t]) / (D[t]+eps) -> out [b][c][t] -
// Per wave: 32(c) x 64(t) tile -> 8 accumulators; B loads shared by 2 A tiles.
__global__ void y_kernel(const __bf16* __restrict__ KPTV,  // [b][C_][M_]
                         const __bf16* __restrict__ QPT,   // [b][M_][T_]
                         const float* __restrict__ Dv,     // [b][T_]
                         float* __restrict__ out) {        // [b][C_][T_]
    const int lane = threadIdx.x & 31;
    const int wave = threadIdx.x >> 5;
    const int lo = lane & 15, hi = lane >> 4;
    const int b = blockIdx.z;
    int job = blockIdx.x * 8 + wave;    // 0..1023
    int c0 = (job >> 6) * 32;           // 16 c-blocks of 32
    int tb = (job & 63) * 64;           // 64 t-chunks of 64
    const __bf16* ar0 = KPTV + ((size_t)b * C_ + c0 + lo) * M_;
    const __bf16* ar1 = ar0 + (size_t)16 * M_;
    const __bf16* QPb = QPT + (size_t)b * M_ * T_;

    v8f acc0[4] = {}, acc1[4] = {};
    for (int k0 = 0; k0 < M_; k0 += 32) {
        v16bf a0 = bf16_a16(ar0 + k0 + hi * 8, ar0 + k0 + 16 + hi * 8);
        v16bf a1 = bf16_a16(ar1 + k0 + hi * 8, ar1 + k0 + 16 + hi * 8);
#pragma unroll
        for (int j = 0; j < 4; ++j) {
            v16bf bm;
#pragma unroll
            for (int s = 0; s < 16; ++s) bm[s] = QPb[(size_t)(k0 + hi * 8 + (s & 7) + (s >> 3) * 16) * T_ + tb + j * 16 + lo];
            acc0[j] = __builtin_amdgcn_wmma_f32_16x16x32_bf16(
                false, a0, false, bm, (short)0, acc0[j], false, false);
            acc1[j] = __builtin_amdgcn_wmma_f32_16x16x32_bf16(
                false, a1, false, bm, (short)0, acc1[j], false, false);
        }
    }
#pragma unroll
    for (int j = 0; j < 4; ++j) {
        int tCol = tb + j * 16 + lo;
        float dinv = 1.f / (Dv[b * T_ + tCol] + EPS_);
#pragma unroll
        for (int i = 0; i < 8; ++i) {
            int cRow = c0 + i + (hi ? 8 : 0);
            out[((size_t)b * C_ + cRow) * T_ + tCol] = acc0[j][i] * dinv;
            out[((size_t)b * C_ + cRow + 16) * T_ + tCol] = acc1[j][i] * dinv;
        }
    }
}

// -------- 6) in-place LayerNorm over t=4096 per (b,c) row of d_out ----------
__global__ void ln_kernel(float* __restrict__ out) {
    __shared__ float r1[256], r2[256];
    float* p = out + (size_t)blockIdx.x * T_;
    float s = 0.f, s2 = 0.f;
    for (int i = threadIdx.x; i < T_; i += 256) {
        float v = p[i]; s += v; s2 += v * v;
    }
    r1[threadIdx.x] = s; r2[threadIdx.x] = s2;
    __syncthreads();
    for (int off = 128; off > 0; off >>= 1) {
        if (threadIdx.x < off) {
            r1[threadIdx.x] += r1[threadIdx.x + off];
            r2[threadIdx.x] += r2[threadIdx.x + off];
        }
        __syncthreads();
    }
    float mu = r1[0] * (1.f / T_);
    float var = r2[0] * (1.f / T_) - mu * mu;
    float rstd = rsqrtf(var + LNEPS_);
    for (int i = threadIdx.x; i < T_; i += 256) p[i] = (p[i] - mu) * rstd;
}

// ---------------------------------------------------------------------------
extern "C" void kernel_launch(void* const* d_in, const int* in_sizes, int n_in,
                              void* d_out, int out_size, void* d_ws, size_t ws_size,
                              hipStream_t stream) {
    const float* q = (const float*)d_in[0];
    const float* k = (const float*)d_in[1];
    const float* v = (const float*)d_in[2];
    const float* W = (const float*)d_in[3];
    float* out = (float*)d_out;

    char* ws = (char*)d_ws;
    __bf16* qpT  = (__bf16*)(ws);                        // [B][M][T] bf16, 16MB
    __bf16* kp   = (__bf16*)(ws + 16777216);             // [B][T][M] bf16, 16MB
    __bf16* kptv = (__bf16*)(ws + 33554432);             // [B][C][M] bf16,  2MB
    float*  ksum = (float*)(ws + 35651584);              // [B][M]
    float*  xdq  = (float*)(ws + 35659776);              // [B][T]
    float*  xdk  = (float*)(ws + 35790848);              // [B][T]
    float*  Dv   = (float*)(ws + 35921920);              // [B][T]
    float*  kpart= (float*)(ws + 36052992);              // [B][16][M] partials

    xd_kernel<<<128, 256, 0, stream>>>(q, xdq);
    xd_kernel<<<128, 256, 0, stream>>>(k, xdk);

    dim3 gf(64, 2, B_);                                  // 64 t-chunks x 16 m-tiles
    feat_kernel<0><<<gf, 256, 0, stream>>>(q, W, xdq, qpT);
    feat_kernel<1><<<gf, 256, 0, stream>>>(k, W, xdk, kp);

    dim3 gks(B_, 16);
    ksum_part_kernel<<<gks, 256, 0, stream>>>(kp, kpart);
    ksum_final_kernel<<<B_, 256, 0, stream>>>(kpart, ksum);
    d_kernel<<<128, 256, 0, stream>>>(qpT, ksum, Dv);

    dim3 gk(8, 1, B_);                                   // 16 c-blocks x 4 m-chunks
    kptv_kernel<<<gk, 256, 0, stream>>>(v, kp, kptv);

    dim3 gy(128, 1, B_);                                 // 16 c-blocks x 64 t-chunks
    y_kernel<<<gy, 256, 0, stream>>>(kptv, qpT, Dv, out);

    ln_kernel<<<B_ * C_, 256, 0, stream>>>(out);
}
